// VQVAE_23184233464495
// MI455X (gfx1250) — matrix operand order repaired and checked by reading, not compiled, not run on hardware
//
#include <hip/hip_runtime.h>
#include <hip/hip_bf16.h>

// ---------------------------------------------------------------------------
// CDNA5 (gfx1250) VQ-VAE forward.  All matrix work (conv, convT, FC, codebook
// distances) runs through v_wmma_f32_16x16x32_f16 (wave32).  Hot loops are
// division-free; weights pre-packed to f16 B-fragment order and streamed into
// LDS with double-buffered GLOBAL_LOAD_ASYNC_TO_LDS_B128 (ASYNCcnt pipelined);
// the A tile is im2col-gathered into LDS in fragment order.
// ---------------------------------------------------------------------------

typedef __attribute__((ext_vector_type(16))) _Float16 v16h;
typedef __attribute__((ext_vector_type(8)))  float    v8f;

struct I2 { int x, y; };

#define CDIV(a,b) (((a)+(b)-1)/(b))

// ---- CDNA5 async global->LDS helpers (inline asm; portable across toolchains)
__device__ __forceinline__ unsigned lds_off32(const void* p)
{
    // generic shared address = {SHARED_BASE, lds_offset}; low 32 bits = offset
    return (unsigned)(unsigned long long)p;
}
__device__ __forceinline__ void async_ld_b128(unsigned lds, const _Float16* g)
{
    asm volatile("global_load_async_to_lds_b128 %0, %1, off"
                 :: "v"(lds), "v"(g) : "memory");
}
__device__ __forceinline__ void wait_async_le2()
{
    asm volatile("s_wait_asynccnt 0x2" ::: "memory");
}
__device__ __forceinline__ void wait_async_0()
{
    asm volatile("s_wait_asynccnt 0x0" ::: "memory");
}

// ---------------------------------------------------------------------------
// Per-layer k-table: for k in [0,Kpad): x = ci*Hin*Win, y = (kh<<8)|kw
// ---------------------------------------------------------------------------
__global__ void __launch_bounds__(256)
ktab_kernel(I2* __restrict__ tab, int Ktot, int Kpad, int KHW, int KW, int HinWin)
{
    const int k = blockIdx.x * blockDim.x + threadIdx.x;
    if (k >= Kpad) return;
    I2 t; t.x = 0; t.y = 0;
    if (k < Ktot) {
        const int ci = k / KHW, rem = k - ci * KHW;
        const int kh = rem / KW, kw = rem - (rem / KW) * KW;
        t.x = ci * HinWin;
        t.y = (kh << 8) | kw;
    }
    tab[k] = t;
}

// ---------------------------------------------------------------------------
// Pack f32 weights into f16 B-fragment order:
//   Wp[((nt*kSteps + ks)*32 + lane)*16 + e],  k = ks*32 + (e&7) + 16*(e>=8) + 8*(lane>=16),
//   col = nt*16 + (lane&15).
//   wmode 0: OIHW | 1: IOHW (convT) | 2: [k][co] dense | 3: [co][k] codebook
// ---------------------------------------------------------------------------
__global__ void __launch_bounds__(256)
pack_weights_kernel(const float* __restrict__ W, _Float16* __restrict__ Wp,
                    int Cout, int Cin, int KH, int KW, int Ktot, int kSteps,
                    int wmode)
{
    const long total = (long)((Cout + 15) >> 4) * kSteps * 512;
    const long i = (long)blockIdx.x * blockDim.x + threadIdx.x;
    if (i >= total) return;
    const int e = (int)(i & 15);
    long t = i >> 4;
    const int lane = (int)(t & 31); t >>= 5;
    const int ks = (int)(t % kSteps);
    const int nt = (int)(t / kSteps);
    const int hlf = lane >> 4, l16 = lane & 15;
    const int k   = (ks << 5) + (e & 7) + ((e & 8) << 1) + (hlf << 3);
    const int col = nt * 16 + l16;
    float v = 0.f;
    if (k < Ktot && col < Cout) {
        const int KHW = KH * KW;
        const int ci = k / KHW, rem = k - ci * KHW;
        const int kh = rem / KW, kw = rem - (rem / KW) * KW;
        long widx;
        if (wmode == 0)      widx = (((long)col * Cin + ci) * KH + kh) * KW + kw;
        else if (wmode == 1) widx = (((long)ci * Cout + col) * KH + kh) * KW + kw;
        else if (wmode == 2) widx = (long)k * Cout + col;
        else                 widx = (long)col * Ktot + k;
        v = W[widx];
    }
    Wp[i] = (_Float16)v;
}

// ---------------------------------------------------------------------------
// Implicit-GEMM conv / convT / dense.  Block = 128 threads (4 waves):
// one 16-row M-tile shared via LDS, 4 N-tiles (one per wave).
// B fragments are double-buffer prefetched into LDS with async loads
// (2x global_load_async_to_lds_b128 per wave per k-step, s_wait_asynccnt
// pipelining); A tile im2col-gathered into LDS in fragment order.
// Steady state per wave per k-step: 2 async issues + ds_load A/B + 1 WMMA.
// ---------------------------------------------------------------------------
template<int WMODE, int STRIDE>
__global__ void __launch_bounds__(128)
conv_wmma_t(const float* __restrict__ X, const _Float16* __restrict__ Wp,
            const float* __restrict__ bias, float* __restrict__ Y,
            const I2* __restrict__ ktab,
            int N, int Cin, int Hin, int Win, int Cout, int Hout, int Wout,
            int pad, int Ktot, int kSteps, int relu)
{
    __shared__ __align__(32) _Float16 As[512];          // A: [lane][e]
    __shared__ __align__(32) _Float16 Bs[2][4][512];    // B: [buf][wave][lane*16+e]
    const int tid  = threadIdx.x;
    const int lane = tid & 31;
    const int wv   = tid >> 5;
    const int hlf  = lane >> 4, l16 = lane & 15;
    const int HWo  = Hout * Wout;
    const long M   = (long)N * HWo;
    const int nT   = (Cout + 15) >> 4;
    const int mt   = blockIdx.x;
    const int nt   = blockIdx.y * 4 + wv;
    const bool nok = nt < nT;

    // ---- gather role (all 128 threads): row = tid>>3, 4 consecutive k ----
    const int  grow16 = tid >> 3;
    const int  gk4    = (tid & 7) << 2;
    const long grow   = (long)mt * 16 + grow16;
    const bool gok    = grow < M;
    const long gm     = gok ? grow : 0;
    const int  gn     = (int)(gm / HWo);
    const int  gsp    = (int)(gm % HWo);
    const int  goh    = gsp / Wout, gow = gsp % Wout;
    const long nbase  = (long)gn * Cin * Hin * Win;

    // ---- compute role ----
    const int  col = nt * 16 + l16;
    const bool cok = nok && (col < Cout);
    const int  ntc = nok ? nt : (nT - 1);           // clamp OOB waves' prefetch
    const _Float16* wp = Wp + (long)ntc * kSteps * 512;
    if (cok) __builtin_prefetch(bias + col, 0, 3);  // global_prefetch_b8

    const unsigned bs0 = lds_off32(&Bs[0][wv][lane * 16]);
    const unsigned bs1 = lds_off32(&Bs[1][wv][lane * 16]);

    // prologue: prefetch k-step 0 into buffer 0 (32B per lane = 2 x b128)
    async_ld_b128(bs0,      wp + lane * 16);
    async_ld_b128(bs0 + 16, wp + lane * 16 + 8);

    v8f acc = {};
    for (int ks = 0; ks < kSteps; ++ks) {
        const unsigned bnxt = (ks & 1) ? bs0 : bs1;
        if (ks + 1 < kSteps) {
            const _Float16* wn = wp + (long)(ks + 1) * 512 + lane * 16;
            async_ld_b128(bnxt,      wn);
            async_ld_b128(bnxt + 16, wn + 8);
            wait_async_le2();   // in-order async: older prefetch has landed
        } else {
            wait_async_0();
        }
        __syncthreads();        // prev wmma A-reads done before A overwrite
        const int kb = ks << 5;
#pragma unroll
        for (int j = 0; j < 4; ++j) {
            const int kl = gk4 + j;
            const int k  = kb + kl;
            const I2 te  = ktab[k];
            const int kh = te.y >> 8, kw = te.y & 255;
            int ih, iw;
            bool ok = gok && (k < Ktot);
            if (WMODE == 1) {           // transposed: x[(oh+p-kh)/s] if exact
                const int ihs = goh + pad - kh, iws = gow + pad - kw;
                if (STRIDE == 2) {
                    ok = ok && ihs >= 0 && iws >= 0 && !(ihs & 1) && !(iws & 1);
                    ih = ihs >> 1; iw = iws >> 1;
                } else {
                    ok = ok && ihs >= 0 && iws >= 0;
                    ih = ihs; iw = iws;
                }
                ok = ok && ih < Hin && iw < Win;
            } else {
                ih = goh * STRIDE - pad + kh;
                iw = gow * STRIDE - pad + kw;
                ok = ok && ih >= 0 && ih < Hin && iw >= 0 && iw < Win;
            }
            // branch-free masked load: clamp address, always load, select
            float v = X[nbase + te.x + (long)(ok ? ih : 0) * Win + (ok ? iw : 0)];
            v = ok ? v : 0.f;
            const int dl = grow16 + ((kl & 8) << 1);      // dest lane
            const int de = (kl & 7) | ((kl & 16) >> 1);   // dest element
            As[dl * 16 + de] = (_Float16)v;
        }
        __syncthreads();
        if (nok) {
            const v16h af = *(const v16h*)(As + lane * 16);
            const v16h bf = *(const v16h*)(Bs[ks & 1][wv] + lane * 16);
            acc = __builtin_amdgcn_wmma_f32_16x16x32_f16(
                      false, af, false, bf, (short)0, acc, false, false);
        }
    }
    if (!cok) return;
    const float bc = bias[col];
#pragma unroll
    for (int r = 0; r < 8; ++r) {
        const long row = (long)mt * 16 + r + (hlf << 3);
        if (row < M) {
            const int n2  = (int)(row / HWo);
            const int sp2 = (int)(row % HWo);
            float v = acc[r] + bc;
            if (relu) v = v > 0.f ? v : 0.f;
            Y[((long)n2 * Cout + col) * HWo + sp2] = v;
        }
    }
}

// ---------------------------------------------------------------------------
// VQ argmin: score(m,j) = |cb_j|^2 - 2*<flat_m, cb_j>  (|f|^2 row-constant).
// Codebook pre-packed to f16 fragment order (kSteps=2, nT=32).
// ---------------------------------------------------------------------------
__global__ void __launch_bounds__(128)
vq_argmin_kernel(const float* __restrict__ lat, const _Float16* __restrict__ cbp,
                 const float* __restrict__ cnorm, int* __restrict__ idxout, int Bm)
{
    const int lane = threadIdx.x & 31;
    const int wv   = threadIdx.x >> 5;
    const int hlf  = lane >> 4, l16 = lane & 15;
    const int mt   = blockIdx.x * 4 + wv;
    const int mT   = (Bm + 15) >> 4;
    if (mt >= mT) return;

    const long mrow = (long)mt * 16 + l16;
    const bool mok  = mrow < Bm;
    const long mm   = mok ? mrow : 0;
    const int  n = (int)(mm / 35), sp = (int)(mm % 35);
    const int  hp = sp / 5, wp = sp % 5;
    const long lbase = ((long)n * 64 * 7 + hp) * 5 + wp;  // + k*35

    float best[8]; int bidx[8];
#pragma unroll
    for (int r = 0; r < 8; ++r) { best[r] = 3.4e38f; bidx[r] = 0; }

    // preload this lane's A fragment (reused across all 32 column tiles)
    v16h afr[2];
#pragma unroll
    for (int ks = 0; ks < 2; ++ks) {
#pragma unroll
        for (int e = 0; e < 16; ++e) {
            const int ka = (e < 8 ? e : 16 + (e - 8)) + (hlf << 3);
            const int k  = (ks << 5) + ka;
            const float av = mok ? lat[lbase + (long)k * 35] : 0.f;
            afr[ks][e] = (_Float16)av;
        }
    }

    for (int ntile = 0; ntile < 32; ++ntile) {
        v8f acc = {};
#pragma unroll
        for (int ks = 0; ks < 2; ++ks) {
            const v16h bf = *(const v16h*)(cbp + (((long)ntile * 2 + ks) * 32 + lane) * 16);
            acc = __builtin_amdgcn_wmma_f32_16x16x32_f16(
                      false, afr[ks], false, bf, (short)0, acc, false, false);
        }
        const int col = ntile * 16 + l16;
        const float cn = cnorm[col];
#pragma unroll
        for (int r = 0; r < 8; ++r) {
            const float d = cn - 2.f * acc[r];
            if (d < best[r]) { best[r] = d; bidx[r] = col; }
        }
    }
#pragma unroll
    for (int r = 0; r < 8; ++r) {
        float v = best[r]; int bi = bidx[r];
        for (int off = 8; off >= 1; off >>= 1) {
            const float ov = __shfl_xor(v, off, 32);
            const int   oi = __shfl_xor(bi, off, 32);
            if (ov < v || (ov == v && oi < bi)) { v = ov; bi = oi; }
        }
        if (l16 == 0) {
            const long row = (long)mt * 16 + r + (hlf << 3);
            if (row < Bm) idxout[row] = bi;
        }
    }
}

// ---------------------------------------------------------------------------
// Support kernels
// ---------------------------------------------------------------------------
__global__ void __launch_bounds__(256)
concat_kernel(const float* __restrict__ ft, const float* __restrict__ fp,
              float* __restrict__ out, long total)
{
    const long i = (long)blockIdx.x * blockDim.x + threadIdx.x;
    if (i >= total) return;
    const int HW = 210 * 160;
    const int hw = (int)(i % HW);
    const long t = i / HW;
    const int c = (int)(t % 9), n = (int)(t / 9);
    const long s = ((long)n * 3 + (c % 3)) * HW + hw;
    const float a = ft[s], b = fp[s];
    out[i] = (c < 3) ? a : ((c < 6) ? b : b - a);
}

__global__ void __launch_bounds__(256)
channel_stats_kernel(const float* __restrict__ X, float* __restrict__ mean,
                     float* __restrict__ var, int N, int C, int HW)
{
    const int c = blockIdx.x;
    const long cnt = (long)N * HW;
    float s = 0.f, s2 = 0.f;
    for (long i = threadIdx.x; i < cnt; i += blockDim.x) {
        const int n = (int)(i / HW), hw = (int)(i % HW);
        const float v = X[((long)n * C + c) * HW + hw];
        s += v; s2 += v * v;
    }
    __shared__ float sh[256], sh2[256];
    sh[threadIdx.x] = s; sh2[threadIdx.x] = s2;
    __syncthreads();
    for (int st = 128; st > 0; st >>= 1) {
        if ((int)threadIdx.x < st) {
            sh[threadIdx.x]  += sh[threadIdx.x + st];
            sh2[threadIdx.x] += sh2[threadIdx.x + st];
        }
        __syncthreads();
    }
    if (threadIdx.x == 0) {
        const float m = sh[0] / (float)cnt;
        mean[c] = m;
        var[c]  = sh2[0] / (float)cnt - m * m;
    }
}

__global__ void __launch_bounds__(256)
bn_relu_kernel(float* __restrict__ X, const float* __restrict__ mean,
               const float* __restrict__ var, const float* __restrict__ g,
               const float* __restrict__ be, int C, int HW, long total)
{
    const long i = (long)blockIdx.x * blockDim.x + threadIdx.x;
    if (i >= total) return;
    const int c = (int)((i / HW) % C);
    const float xh = (X[i] - mean[c]) * rsqrtf(var[c] + 1e-5f);
    const float v = xh * g[c] + be[c];
    X[i] = v > 0.f ? v : 0.f;
}

__global__ void __launch_bounds__(256)
film_kernel(float* __restrict__ X, const float* __restrict__ f, int FCW,
            int goff, int boff, int C, int HW, long total)
{
    const long i = (long)blockIdx.x * blockDim.x + threadIdx.x;
    if (i >= total) return;
    const int n = (int)(i / ((long)C * HW));
    const int c = (int)((i / HW) % C);
    X[i] = f[(long)n * FCW + goff + c] * X[i] + f[(long)n * FCW + boff + c];
}

__global__ void __launch_bounds__(256)
spatial_mean_kernel(const float* __restrict__ X, float* __restrict__ feat,
                    int NC, int HW)
{
    const int i = blockIdx.x * blockDim.x + threadIdx.x;
    if (i >= NC) return;
    float s = 0.f;
    for (int j = 0; j < HW; ++j) s += X[(long)i * HW + j];
    feat[i] = s / (float)HW;
}

__global__ void __launch_bounds__(256)
cnorm_kernel(const float* __restrict__ cb, float* __restrict__ cn)
{
    const int j = blockIdx.x * blockDim.x + threadIdx.x;
    if (j >= 512) return;
    float s = 0.f;
    for (int k = 0; k < 64; ++k) { const float v = cb[j * 64 + k]; s += v * v; }
    cn[j] = s;
}

__global__ void __launch_bounds__(256)
zero_kernel(float* __restrict__ p, long n)
{
    const long i = (long)blockIdx.x * blockDim.x + threadIdx.x;
    if (i < n) p[i] = 0.f;
}

__global__ void __launch_bounds__(256)
quant_loss_kernel(const float* __restrict__ lat, const float* __restrict__ cb,
                  const int* __restrict__ idx, float* __restrict__ quant,
                  float* __restrict__ sumsq)
{
    const long i = (long)blockIdx.x * blockDim.x + threadIdx.x; // over 71680
    __shared__ float sh[256];
    float d2 = 0.f;
    if (i < 71680) {
        const int m = (int)(i >> 6), dd = (int)(i & 63);
        const int n = m / 35, sp = m % 35, hp = sp / 5, wp = sp % 5;
        const long li = (((long)n * 64 + dd) * 7 + hp) * 5 + wp;
        const float q = cb[(long)idx[m] * 64 + dd];
        quant[li] = q;
        const float df = lat[li] - q;
        d2 = df * df;
    }
    sh[threadIdx.x] = d2;
    __syncthreads();
    for (int st = 128; st > 0; st >>= 1) {
        if ((int)threadIdx.x < st) sh[threadIdx.x] += sh[threadIdx.x + st];
        __syncthreads();
    }
    if (threadIdx.x == 0) atomicAdd(sumsq, sh[0]);
}

__global__ void __launch_bounds__(256)
counts_kernel(const int* __restrict__ idx, float* __restrict__ counts)
{
    const int i = blockIdx.x * blockDim.x + threadIdx.x;
    if (i < 1120) atomicAdd(&counts[idx[i]], 1.f);
}

__global__ void __launch_bounds__(512)
finalize_kernel(const float* __restrict__ counts, const float* __restrict__ sumsq,
                const int* __restrict__ idx, float* __restrict__ out)
{
    __shared__ float sh[512];
    const int t = threadIdx.x;
    const float avg = counts[t] / 1120.f;
    sh[t] = -avg * logf(avg + 1e-10f);
    __syncthreads();
    for (int st = 256; st > 0; st >>= 1) {
        if (t < st) sh[t] += sh[t + st];
        __syncthreads();
    }
    const long R = 3225600;
    if (t == 0) {
        out[R]     = 0.25f * sumsq[0] / 71680.f;
        out[R + 1] = expf(sh[0]);
    }
    for (int i = t; i < 1120; i += 512) out[R + 2 + i] = (float)idx[i];
}

__global__ void __launch_bounds__(256)
sigmoid_kernel(const float* __restrict__ X, float* __restrict__ out, long n)
{
    const long i = (long)blockIdx.x * blockDim.x + threadIdx.x;
    if (i < n) out[i] = 1.f / (1.f + expf(-X[i]));
}

// ---------------------------------------------------------------------------
// Host-side helpers
// ---------------------------------------------------------------------------
static void run_conv(const float* X, const float* W, const float* b, float* Y,
                     int N, int Cin, int Hin, int Win, int Cout, int Hout,
                     int Wout, int KH, int KW, int stride, int pad, int wmode,
                     int relu, I2* ktab, _Float16* Wp, hipStream_t st)
{
    const int KHW = KH * KW, Ktot = Cin * KHW;
    const int kSteps = CDIV(Ktot, 32), Kpad = kSteps * 32;
    ktab_kernel<<<dim3(CDIV(Kpad, 256)), dim3(256), 0, st>>>(
        ktab, Ktot, Kpad, KHW, KW, Hin * Win);
    const int nT = CDIV(Cout, 16);
    const long ptot = (long)nT * kSteps * 512;
    pack_weights_kernel<<<dim3((int)CDIV(ptot, 256)), dim3(256), 0, st>>>(
        W, Wp, Cout, Cin, KH, KW, Ktot, kSteps, wmode);
    const long M = (long)N * Hout * Wout;
    dim3 g((unsigned)CDIV(M, 16), (unsigned)CDIV(nT, 4));
#define CARGS X, Wp, b, Y, ktab, N, Cin, Hin, Win, Cout, Hout, Wout, pad, Ktot, kSteps, relu
    if (wmode == 1)       conv_wmma_t<1, 2><<<g, dim3(128), 0, st>>>(CARGS);
    else if (wmode == 2)  conv_wmma_t<2, 1><<<g, dim3(128), 0, st>>>(CARGS);
    else if (stride == 2) conv_wmma_t<0, 2><<<g, dim3(128), 0, st>>>(CARGS);
    else                  conv_wmma_t<0, 1><<<g, dim3(128), 0, st>>>(CARGS);
#undef CARGS
}

static void launch_bn(float* X, float* mean, float* var, const float* g,
                      const float* be, int N, int C, int HW, hipStream_t st)
{
    channel_stats_kernel<<<dim3(C), dim3(256), 0, st>>>(X, mean, var, N, C, HW);
    const long total = (long)N * C * HW;
    bn_relu_kernel<<<dim3((int)CDIV(total, 256)), dim3(256), 0, st>>>(
        X, mean, var, g, be, C, HW, total);
}

extern "C" void kernel_launch(void* const* d_in, const int* in_sizes, int n_in,
                              void* d_out, int out_size, void* d_ws, size_t ws_size,
                              hipStream_t stream)
{
    (void)in_sizes; (void)n_in; (void)out_size; (void)ws_size;
    auto F = [&](int i) { return (const float*)d_in[i]; };
    const float* frame_t  = F(0);
    const float* frame_tp = F(1);
    float* out = (float*)d_out;

    // ---- workspace arena (floats; all offsets multiples of 4 -> 16B aligned)
    const long BUF = 17203200;  // 32*64*105*80 (largest activation)
    float* ws    = (float*)d_ws;
    float* buf0  = ws;
    float* bufA  = buf0 + BUF;
    float* lat   = bufA + BUF;         // 71680
    float* quant = lat + 71680;        // 71680
    float* feat  = quant + 71680;      // 4096
    float* f1    = feat + 4096;        // 32768
    float* f3    = f1 + 32768;         // 16384
    float* meanb = f3 + 16384;         // 512
    float* varb  = meanb + 512;        // 512
    float* cnb   = varb + 512;         // 512
    float* cnts  = cnb + 512;          // 512
    float* ssq   = cnts + 512;         // 16 (1 used)
    int*   idxb  = (int*)(ssq + 16);   // 1120 ints (1136 slots)
    _Float16* Wp  = (_Float16*)(ssq + 16 + 1136);  // 2,359,296 halves max
    _Float16* cbp = Wp + 2359296;                  // 32768 halves
    I2*       ktab = (I2*)(cbp + 32768);           // 8192 I2 (64KB)

    // ---- encoder ----
    {
        const long total = 32L * 9 * 210 * 160;
        concat_kernel<<<dim3((int)CDIV(total, 256)), dim3(256), 0, stream>>>(
            frame_t, frame_tp, buf0, total);
    }
    const int ech[6] = {9, 64, 128, 256, 512, 512};
    const int ekk[5] = {4, 4, 4, 4, 3};
    const int eH[6]  = {210, 105, 52, 26, 13, 7};
    const int eW[6]  = {160, 80, 40, 20, 10, 5};
    for (int i = 0; i < 5; ++i) {
        float* in = (i % 2 == 0) ? buf0 : bufA;
        float* o  = (i % 2 == 0) ? bufA : buf0;
        run_conv(in, F(2 + 4 * i), F(3 + 4 * i), o, 32, ech[i], eH[i], eW[i],
                 ech[i + 1], eH[i + 1], eW[i + 1], ekk[i], ekk[i], 2, 1, 0, 0,
                 ktab, Wp, stream);
        launch_bn(o, meanb, varb, F(4 + 4 * i), F(5 + 4 * i), 32, ech[i + 1],
                  eH[i + 1] * eW[i + 1], stream);
    }
    // 1x1 projection to D=64 (reads bufA = e4 output)
    run_conv(bufA, F(22), F(23), lat, 32, 512, 7, 5, 64, 7, 5, 1, 1, 1, 0,
             0, 0, ktab, Wp, stream);

    // ---- vector quantization ----
    cnorm_kernel<<<dim3(2), dim3(256), 0, stream>>>(F(24), cnb);
    {   // pack codebook rows [col][k] -> fragment order (Cout=512, K=64)
        const long ptot = 32L * 2 * 512;
        pack_weights_kernel<<<dim3((int)CDIV(ptot, 256)), dim3(256), 0, stream>>>(
            F(24), cbp, 512, 64, 1, 1, 64, 2, 3);
    }
    vq_argmin_kernel<<<dim3(CDIV(70, 4)), dim3(128), 0, stream>>>(
        lat, cbp, cnb, idxb, 1120);
    zero_kernel<<<dim3(3), dim3(256), 0, stream>>>(cnts, 528);
    quant_loss_kernel<<<dim3(CDIV(71680, 256)), dim3(256), 0, stream>>>(
        lat, F(24), idxb, quant, ssq);
    counts_kernel<<<dim3(CDIV(1120, 256)), dim3(256), 0, stream>>>(idxb, cnts);

    // ---- conditioning net (frame_t) ----
    const int cch[4] = {3, 32, 64, 128};
    const int cH[4]  = {210, 105, 53, 27};
    const int cW[4]  = {160, 80, 40, 20};
    {
        float* cin[3]  = {nullptr, bufA, buf0};
        float* cout[3] = {bufA, buf0, bufA};
        for (int i = 0; i < 3; ++i) {
            const float* in = (i == 0) ? frame_t : cin[i];
            run_conv(in, F(25 + 2 * i), F(26 + 2 * i), cout[i], 32, cch[i],
                     cH[i], cW[i], cch[i + 1], cH[i + 1], cW[i + 1], 5, 5, 2, 2,
                     0, 1, ktab, Wp, stream);
        }
    }
    spatial_mean_kernel<<<dim3(CDIV(32 * 128, 256)), dim3(256), 0, stream>>>(
        bufA, feat, 32 * 128, 27 * 20);
    // FC layers (dense wmode=2: weights [K][Cout])
    run_conv(feat, F(31), F(32), f1, 32, 128, 1, 1, 1024, 1, 1, 1, 1, 1, 0,
             2, 0, ktab, Wp, stream);
    run_conv(feat, F(33), F(34), f3, 32, 128, 1, 1, 512, 1, 1, 1, 1, 1, 0,
             2, 0, ktab, Wp, stream);

    // ---- decoder (transposed convs, wmode=1) ----
    const int tci[5] = {64, 512, 256, 128, 64};
    const int tco[5] = {512, 256, 128, 64, 3};
    const int tk[5]  = {3, 4, 4, 4, 4};
    const int dHi[5] = {7, 13, 26, 52, 105};
    const int dWi[5] = {5, 10, 20, 40, 80};
    const int dHo[5] = {13, 26, 52, 105, 210};
    const int dWo[5] = {10, 20, 40, 80, 160};
    float* din[5]  = {quant, bufA, buf0, bufA, buf0};
    float* dout[5] = {bufA, buf0, bufA, buf0, bufA};
    for (int i = 0; i < 5; ++i) {
        run_conv(din[i], F(35 + 2 * i), F(36 + 2 * i), dout[i], 32, tci[i],
                 dHi[i], dWi[i], tco[i], dHo[i], dWo[i], tk[i], tk[i], 2, 1,
                 1, 0, ktab, Wp, stream);
        const int HW = dHo[i] * dWo[i];
        const long total = 32L * tco[i] * HW;
        if (i == 0) {
            film_kernel<<<dim3((int)CDIV(total, 256)), dim3(256), 0, stream>>>(
                dout[i], f1, 1024, 0, 512, 512, HW, total);
        } else if (i == 1) {
            film_kernel<<<dim3((int)CDIV(total, 256)), dim3(256), 0, stream>>>(
                dout[i], f3, 512, 0, 256, 256, HW, total);
        }
        if (i < 4) {
            launch_bn(dout[i], meanb, varb, F(45 + 2 * i), F(46 + 2 * i), 32,
                      tco[i], HW, stream);
        }
    }
    // recon = sigmoid(t4 output)
    {
        const long n = 32L * 3 * 210 * 160;
        sigmoid_kernel<<<dim3((int)CDIV(n, 256)), dim3(256), 0, stream>>>(
            bufA, out, n);
    }
    finalize_kernel<<<dim3(1), dim3(512), 0, stream>>>(cnts, ssq, idxb, out);
}
